// MultiHeadAttention_59940563583611
// MI455X (gfx1250) — matrix operand order repaired and checked by reading, not compiled, not run on hardware
//
#include <hip/hip_runtime.h>

// ---------------------------------------------------------------------------
// MHA forward for MI455X (gfx1250), wave32 + v_wmma_f32_16x16x32_f16.
// All GEMM operands staged once as f16: inner loops are pure
// global_load_b128 + v_wmma. Per k-step: 2 A frags + 4 B frags loaded up
// front (one clause), then 8 independent WMMAs. The 1-GiB softmax-weights
// operand of the PV GEMM is prefetched one k-step ahead (global_prefetch_b8).
//
// d_out = [ out: 4*2048*1024 f32 | weights: 4*16*2048*2048 f32 ]
// d_ws  = [ qh 16MB | kh 16MB | vT 16MB | Xbuf/Hb 16MB | Wbuf 2MB ]  (f16)
// ---------------------------------------------------------------------------

typedef __attribute__((ext_vector_type(16))) _Float16 v16h;
typedef __attribute__((ext_vector_type(8)))  _Float16 v8h;
typedef __attribute__((ext_vector_type(4)))  _Float16 v4h;
typedef __attribute__((ext_vector_type(8)))  float    v8f;

#define HEADS 16
#define HDIM  64
#define DIM   1024
#define SEQ   2048
#define BATCH 4
#define NEGV  -1000000000.0f

// ---- WMMA fragment loaders (ISA 7.12.2 layouts, wave32) -------------------
// A 16x32 f16: lane L holds row M=L&15; two 8-elem runs at k0+hs*8, k0+16+hs*8.
__device__ __forceinline__ v16h a_frag_f16(const _Float16* __restrict__ p, int hs) {
  v8h lo = *(const v8h*)(p + hs * 8);
  v8h hi = *(const v8h*)(p + 16 + hs * 8);
  v16h a;
#pragma unroll
  for (int e = 0; e < 8; ++e) { a[e] = lo[e]; a[e + 8] = hi[e]; }
  return a;
}

// A from f32 source (only needed for the softmax-probability operand).
__device__ __forceinline__ v16h a_frag_f32(const float* __restrict__ p, int hs) {
  v16h a;
  const float* q0 = p + hs * 8;
#pragma unroll
  for (int e = 0; e < 8; ++e) a[e]     = (_Float16)q0[e];
#pragma unroll
  for (int e = 0; e < 8; ++e) a[e + 8] = (_Float16)q0[16 + e];
  return a;
}

// B 32x16 f16: lane L holds column N=L&15; two 8-elem runs at k0+hs*8, k0+16+hs*8
// within row N of the K-major source (B = source^T).
__device__ __forceinline__ v16h b_frag_f16(const _Float16* __restrict__ p, int hs) {
  v8h lo = *(const v8h*)(p + hs * 8);
  v8h hi = *(const v8h*)(p + 16 + hs * 8);
  v16h b;
#pragma unroll
  for (int e = 0; e < 8; ++e) { b[e] = lo[e]; b[e + 8] = hi[e]; }
  return b;
}

#define WMMA_F16(A, B, C) \
  __builtin_amdgcn_wmma_f32_16x16x32_f16(false, (A), false, (B), (short)0, (C), false, false)

// ---------------------------------------------------------------------------
// Kernel 0: one-shot f32 -> f16 staging (n multiple of 1024).
// ---------------------------------------------------------------------------
__global__ void cvt_kernel(const float* __restrict__ src, _Float16* __restrict__ dst) {
  const size_t i = ((size_t)blockIdx.x * blockDim.x + threadIdx.x) * 4;
  const float4 v = *(const float4*)(src + i);
  v4h o;
  o[0] = (_Float16)v.x; o[1] = (_Float16)v.y;
  o[2] = (_Float16)v.z; o[3] = (_Float16)v.w;
  *(v4h*)(dst + i) = o;
}

// ---------------------------------------------------------------------------
// Kernel 1: projection  dst = X @ W^T  (8192x1024)@(1024x1024)^T, f16 in/out.
// One wave -> 32x64 tile. mode 0: dst[bh][s][d]   (q,k)
//                         mode 2: dst[bh][d][s]   (v transposed for PV GEMM)
// ---------------------------------------------------------------------------
__global__ void proj_kernel(const _Float16* __restrict__ X,
                            const _Float16* __restrict__ W,
                            _Float16* __restrict__ dst, int mode) {
  const int lane = threadIdx.x;
  const int mlo = lane & 15, hs = lane >> 4;
  const int rt = blockIdx.x;        // 256 row tiles of 32
  const int h  = blockIdx.y;        // 16 col tiles of 64 == head index

  const _Float16* arow0 = X + ((size_t)(rt * 32 + mlo)) * DIM;
  const _Float16* arow1 = arow0 + (size_t)16 * DIM;
  const _Float16* brow[4];
#pragma unroll
  for (int t = 0; t < 4; ++t)
    brow[t] = W + ((size_t)(h * 64 + t * 16 + mlo)) * DIM;

  v8f acc[2][4] = {};
#pragma unroll 2
  for (int k0 = 0; k0 < DIM; k0 += 32) {
    // Load all fragments for this k-step first (independent -> one clause).
    v16h a0 = a_frag_f16(arow0 + k0, hs);
    v16h a1 = a_frag_f16(arow1 + k0, hs);
    v16h bb[4];
#pragma unroll
    for (int t = 0; t < 4; ++t) bb[t] = b_frag_f16(brow[t] + k0, hs);
    // Then 8 independent WMMAs.
#pragma unroll
    for (int t = 0; t < 4; ++t) {
      acc[0][t] = WMMA_F16(a0, bb[t], acc[0][t]);
      acc[1][t] = WMMA_F16(a1, bb[t], acc[1][t]);
    }
  }

  const int b  = rt >> 6;           // 64 row-tiles (of 32) per batch
  const int s0 = (rt & 63) * 32;
  const int bh = b * HEADS + h;
#pragma unroll
  for (int half = 0; half < 2; ++half) {
#pragma unroll
    for (int j = 0; j < 8; ++j) {
      const int s = s0 + half * 16 + j + 8 * hs;   // M = j + 8*(lane>=16)
#pragma unroll
      for (int t = 0; t < 4; ++t) {
        const int d = t * 16 + mlo;
        const _Float16 v = (_Float16)acc[half][t][j];
        if (mode == 2) dst[((size_t)bh * HDIM + d) * SEQ + s] = v;  // vT[bh][d][s]
        else           dst[((size_t)bh * SEQ + s) * HDIM + d] = v;  // q/k[bh][s][d]
      }
    }
  }
}

// ---------------------------------------------------------------------------
// Kernel 2: scores[bh][q][k] = (q.k)/8 with causal mask (-1e9 above diag).
// One wave -> 32x16 tile (B reused by 2 A fragments); fully-masked tiles
// skip the GEMM and just store NEG (softmax turns them into exact zeros).
// ---------------------------------------------------------------------------
__global__ void score_kernel(const _Float16* __restrict__ qh,
                             const _Float16* __restrict__ kh,
                             float* __restrict__ Wt) {
  const int lane = threadIdx.x;
  const int mlo = lane & 15, hs = lane >> 4;
  const int qt = blockIdx.x;        // 64 q-tiles of 32 rows
  const int kt = blockIdx.y;        // 128 k-tiles of 16 cols
  const int bh = blockIdx.z;

  v8f acc[2] = {};
  if (kt <= 2 * qt + 1) {                           // uniform: tile touches causal region
    const _Float16* qrow0 = qh + ((size_t)bh * SEQ + qt * 32 + mlo) * HDIM;
    const _Float16* qrow1 = qrow0 + (size_t)16 * HDIM;
    const _Float16* krow  = kh + ((size_t)bh * SEQ + kt * 16 + mlo) * HDIM;
    // Both k-steps' fragments loaded up front (HDIM = 64 = 2 steps).
    v16h a00 = a_frag_f16(qrow0 +  0, hs), a10 = a_frag_f16(qrow1 +  0, hs);
    v16h a01 = a_frag_f16(qrow0 + 32, hs), a11 = a_frag_f16(qrow1 + 32, hs);
    v16h b0  = b_frag_f16(krow  +  0, hs);
    v16h b1  = b_frag_f16(krow  + 32, hs);
    acc[0] = WMMA_F16(a00, b0, acc[0]);
    acc[1] = WMMA_F16(a10, b0, acc[1]);
    acc[0] = WMMA_F16(a01, b1, acc[0]);
    acc[1] = WMMA_F16(a11, b1, acc[1]);
  }

  float* wrow = Wt + (size_t)bh * SEQ * SEQ;
  const int n = kt * 16 + mlo;
#pragma unroll
  for (int half = 0; half < 2; ++half) {
#pragma unroll
    for (int j = 0; j < 8; ++j) {
      const int m = qt * 32 + half * 16 + j + 8 * hs;
      const float v = (n <= m) ? acc[half][j] * 0.125f : NEGV;
      wrow[(size_t)m * SEQ + n] = v;
    }
  }
}

// ---------------------------------------------------------------------------
// Kernel 3: in-place row softmax over 2048 entries. One 256-thread block/row.
// ---------------------------------------------------------------------------
__global__ void softmax_kernel(float* __restrict__ Wt) {
  __shared__ float buf[SEQ];
  __shared__ float red[256];
  float* row = Wt + (size_t)blockIdx.x * SEQ;
  const int tid = threadIdx.x;

  float mx = -3.0e38f;
  for (int i = tid; i < SEQ; i += 256) { float v = row[i]; buf[i] = v; mx = fmaxf(mx, v); }
  red[tid] = mx; __syncthreads();
  for (int s = 128; s > 0; s >>= 1) { if (tid < s) red[tid] = fmaxf(red[tid], red[tid + s]); __syncthreads(); }
  mx = red[0]; __syncthreads();

  float sm = 0.0f;
  for (int i = tid; i < SEQ; i += 256) { float e = __expf(buf[i] - mx); buf[i] = e; sm += e; }
  red[tid] = sm; __syncthreads();
  for (int s = 128; s > 0; s >>= 1) { if (tid < s) red[tid] += red[tid + s]; __syncthreads(); }
  const float inv = 1.0f / red[0];

  for (int i = tid; i < SEQ; i += 256) row[i] = buf[i] * inv;
}

// ---------------------------------------------------------------------------
// Kernel 4: H[b][s][h*64+d] = P @ V.  One wave -> 32x64 tile.
// Causal: q-tile qt (32 rows) only needs qt+1 K-steps of 32; P is exactly 0
// above the diagonal so padded reads are harmless. P streams ~1 GiB from
// HBM, so the next k-step's rows are prefetched (global_prefetch_b8).
// ---------------------------------------------------------------------------
__global__ void av_kernel(const float* __restrict__ Wt,
                          const _Float16* __restrict__ vT,
                          _Float16* __restrict__ H) {
  const int lane = threadIdx.x;
  const int mlo = lane & 15, hs = lane >> 4;
  const int qt = blockIdx.x;        // 64 q-tiles of 32 rows
  const int bh = blockIdx.y;

  const float* prow0 = Wt + (size_t)bh * SEQ * SEQ + (size_t)(qt * 32 + mlo) * SEQ;
  const float* prow1 = prow0 + (size_t)16 * SEQ;
  const _Float16* vb = vT + (size_t)bh * HDIM * SEQ;

  v8f acc[2][4] = {};
  const int nsteps = qt + 1;
  for (int kc = 0; kc < nsteps; ++kc) {
    const int k0 = kc * 32;
    if (kc + 1 < nsteps) {          // prefetch next k-step of the HBM stream
      __builtin_prefetch(prow0 + k0 + 32, 0, 1);
      __builtin_prefetch(prow1 + k0 + 32, 0, 1);
    }
    v16h a0 = a_frag_f32(prow0 + k0, hs);
    v16h a1 = a_frag_f32(prow1 + k0, hs);
    v16h bb[4];
#pragma unroll
    for (int t = 0; t < 4; ++t)
      bb[t] = b_frag_f16(vb + (size_t)(t * 16 + mlo) * SEQ + k0, hs); // B[kk][n]=vT[n][kk]
#pragma unroll
    for (int t = 0; t < 4; ++t) {
      acc[0][t] = WMMA_F16(a0, bb[t], acc[0][t]);
      acc[1][t] = WMMA_F16(a1, bb[t], acc[1][t]);
    }
  }

  const int b = bh >> 4, h = bh & 15;
#pragma unroll
  for (int half = 0; half < 2; ++half) {
#pragma unroll
    for (int j = 0; j < 8; ++j) {
      const int s = qt * 32 + half * 16 + j + 8 * hs;
      const size_t off = ((size_t)b * SEQ + s) * DIM + h * HDIM;
#pragma unroll
      for (int t = 0; t < 4; ++t) H[off + t * 16 + mlo] = (_Float16)acc[half][t][j];
    }
  }
}

// ---------------------------------------------------------------------------
// Kernel 5: out = H @ Wo^T + bo  -> d_out (f32).  One wave -> 32x64 tile.
// ---------------------------------------------------------------------------
__global__ void oproj_kernel(const _Float16* __restrict__ H,
                             const _Float16* __restrict__ Wo,
                             const float* __restrict__ bo,
                             float* __restrict__ out) {
  const int lane = threadIdx.x;
  const int mlo = lane & 15, hs = lane >> 4;
  const int rt = blockIdx.x;      // 256 row tiles of 32
  const int ct = blockIdx.y;      // 16 col tiles of 64

  const _Float16* arow0 = H + (size_t)(rt * 32 + mlo) * DIM;
  const _Float16* arow1 = arow0 + (size_t)16 * DIM;
  const _Float16* brow[4];
#pragma unroll
  for (int t = 0; t < 4; ++t)
    brow[t] = Wo + (size_t)(ct * 64 + t * 16 + mlo) * DIM;

  v8f acc[2][4] = {};
#pragma unroll 2
  for (int k0 = 0; k0 < DIM; k0 += 32) {
    v16h a0 = a_frag_f16(arow0 + k0, hs);
    v16h a1 = a_frag_f16(arow1 + k0, hs);
    v16h bb[4];
#pragma unroll
    for (int t = 0; t < 4; ++t) bb[t] = b_frag_f16(brow[t] + k0, hs);
#pragma unroll
    for (int t = 0; t < 4; ++t) {
      acc[0][t] = WMMA_F16(a0, bb[t], acc[0][t]);
      acc[1][t] = WMMA_F16(a1, bb[t], acc[1][t]);
    }
  }

#pragma unroll
  for (int half = 0; half < 2; ++half) {
#pragma unroll
    for (int j = 0; j < 8; ++j) {
      const int row = rt * 32 + half * 16 + j + 8 * hs;
#pragma unroll
      for (int t = 0; t < 4; ++t) {
        const int col = ct * 64 + t * 16 + mlo;
        out[(size_t)row * DIM + col] = acc[half][t][j] + bo[col];
      }
    }
  }
}

// ---------------------------------------------------------------------------
extern "C" void kernel_launch(void* const* d_in, const int* in_sizes, int n_in,
                              void* d_out, int out_size, void* d_ws, size_t ws_size,
                              hipStream_t stream) {
  const float* Q  = (const float*)d_in[0];
  const float* K  = (const float*)d_in[1];
  const float* V  = (const float*)d_in[2];
  const float* Wq = (const float*)d_in[3];
  const float* Wk = (const float*)d_in[4];
  const float* Wv = (const float*)d_in[5];
  const float* Wo = (const float*)d_in[6];
  const float* bo = (const float*)d_in[7];

  float* out     = (float*)d_out;
  float* weights = out + (size_t)BATCH * SEQ * DIM;      // second tuple output

  const size_t NX = (size_t)BATCH * SEQ * DIM;           // 8M activation elems
  const size_t NW = (size_t)DIM * DIM;                   // 1M weight elems
  _Float16* qh   = (_Float16*)d_ws;
  _Float16* kh   = qh + NX;
  _Float16* vT   = kh + NX;
  _Float16* Xbuf = vT + NX;                              // staging; later reused as Hb
  _Float16* Wbuf = Xbuf + NX;
  _Float16* Hb   = Xbuf;

  const int XB = (int)(NX / 1024);                       // cvt grid sizes
  const int WB = (int)(NW / 1024);

  // Q/K/V projections: stage activations + weights as f16, then WMMA GEMM.
  cvt_kernel<<<XB, 256, 0, stream>>>(Q, Xbuf);
  cvt_kernel<<<WB, 256, 0, stream>>>(Wq, Wbuf);
  proj_kernel<<<dim3(256, 16), 32, 0, stream>>>(Xbuf, Wbuf, qh, 0);

  cvt_kernel<<<XB, 256, 0, stream>>>(K, Xbuf);
  cvt_kernel<<<WB, 256, 0, stream>>>(Wk, Wbuf);
  proj_kernel<<<dim3(256, 16), 32, 0, stream>>>(Xbuf, Wbuf, kh, 0);

  cvt_kernel<<<XB, 256, 0, stream>>>(V, Xbuf);
  cvt_kernel<<<WB, 256, 0, stream>>>(Wv, Wbuf);
  proj_kernel<<<dim3(256, 16), 32, 0, stream>>>(Xbuf, Wbuf, vT, 2);

  // Attention.
  score_kernel<<<dim3(64, 128, BATCH * HEADS), 32, 0, stream>>>(qh, kh, weights);
  softmax_kernel<<<dim3(BATCH * HEADS * SEQ), 256, 0, stream>>>(weights);
  av_kernel<<<dim3(64, BATCH * HEADS), 32, 0, stream>>>(weights, vT, Hb);

  // Output projection.
  cvt_kernel<<<WB, 256, 0, stream>>>(Wo, Wbuf);
  oproj_kernel<<<dim3(256, 16), 32, 0, stream>>>(Hb, Wbuf, bo, out);
}
